// GCN_75874892251588
// MI455X (gfx1250) — matrix operand order repaired and checked by reading, not compiled, not run on hardware
//
#include <hip/hip_runtime.h>
#include <hip/hip_bf16.h>

typedef __attribute__((ext_vector_type(16))) __bf16 v16bf;
typedef __attribute__((ext_vector_type(8)))  __bf16 v8bf;
typedef __attribute__((ext_vector_type(8)))  float  v8f;

#define GN 50000
#define GE 800000
#define GH 128
#define GC 10

// ---------------- degree / dinv ----------------
__global__ void zero_f32(float* __restrict__ p, int n) {
  int i = blockIdx.x * blockDim.x + threadIdx.x;
  if (i < n) p[i] = 0.0f;
}

__global__ void degree_kernel(const int* __restrict__ dst, float* __restrict__ deg, int ne) {
  int e = blockIdx.x * blockDim.x + threadIdx.x;
  if (e < ne) atomicAdd(&deg[dst[e]], 1.0f);
}

__global__ void dinv_kernel(const float* __restrict__ deg, float* __restrict__ dinv, int n) {
  int i = blockIdx.x * blockDim.x + threadIdx.x;
  if (i < n) dinv[i] = rsqrtf(deg[i] + 1.0f);
}

// ---------------- one-time input convert: Xb = bf16(x) ----------------
__global__ void convert_x_kernel(const float* __restrict__ x, __bf16* __restrict__ Xb, int n) {
  int i = blockIdx.x * blockDim.x + threadIdx.x;
  if (i < n) Xb[i] = (__bf16)x[i];
}

// ---------------- weight pack: B-fragment lane order ----------------
// Wp[ ((kt*128 + col)*32) + half*16 + e ] = bf16( W[(kt*32 + half*16 + e)*128 + col] )
__global__ void pack_w_kernel(const float* __restrict__ W, __bf16* __restrict__ Wp) {
  int idx = blockIdx.x * blockDim.x + threadIdx.x;   // 16384 elements
  int e    = idx & 15;
  int half = (idx >> 4) & 1;
  int col  = (idx >> 5) & 127;
  int kt   = idx >> 12;
  int k    = kt * 32 + half * 16 + e;
  Wp[idx] = (__bf16)W[(size_t)k * GH + col];
}

// output weights 128x10, zero-padded to 16 cols:
// Wp[ kt*512 + col*32 + half*16 + e ], col in [0,16)
__global__ void pack_wo_kernel(const float* __restrict__ Wo, __bf16* __restrict__ Wp) {
  int idx = blockIdx.x * blockDim.x + threadIdx.x;   // 2048 elements
  int e    = idx & 15;
  int half = (idx >> 4) & 1;
  int col  = (idx >> 5) & 15;
  int kt   = idx >> 9;
  int k    = kt * 32 + half * 16 + e;
  Wp[idx] = (col < GC) ? (__bf16)Wo[(size_t)k * GC + col] : (__bf16)0.0f;
}

// ---------------- WMMA GEMM: T(f32) = Xb(N,128 bf16) @ Wp(packed bf16) ----------------
// grid.x = N/16 row tiles; block = 256 = 8 waves; wave w -> column tile w.
__global__ void gemm_wmma(const __bf16* __restrict__ Xb, const __bf16* __restrict__ Wp,
                          float* __restrict__ T) {
  const int wave = threadIdx.x >> 5;
  const int lane = threadIdx.x & 31;
  const int m    = lane & 15;
  const int half = lane >> 4;
  const int rt   = blockIdx.x;
  const int col  = wave * 16 + m;

  const __bf16* xr0 = Xb + (size_t)(rt * 16 + m) * GH;
  const __bf16* wp0 = Wp + (size_t)col * 32 + half * 16;
  v8f acc = {};
#pragma unroll
  for (int kt = 0; kt < 4; ++kt) {
    const __bf16* xr = xr0 + kt * 32 + half * 16;
    v8bf alo = *(const v8bf*)xr;          // K = kt*32 + half*16 + [0..8)
    v8bf ahi = *(const v8bf*)(xr + 8);    // K = kt*32 + half*16 + 8 + [0..8)
    v16bf a;
#pragma unroll
    for (int e = 0; e < 8; ++e) { a[e] = alo[e]; a[e + 8] = ahi[e]; }
    v16bf b = *(const v16bf*)(wp0 + (size_t)kt * (GH * 32));  // K = kt*32+half*16+e
    acc = __builtin_amdgcn_wmma_f32_16x16x32_bf16(false, a, false, b,
                                                  (short)0, acc, false, false);
  }
  float* out = T + (size_t)(rt * 16 + half * 8) * GH + col;
#pragma unroll
  for (int g = 0; g < 8; ++g) out[(size_t)g * GH] = acc[g];
}

// ---------------- self-loop init: Agg = T * dinv^2 ----------------
__global__ void self_loop_kernel(const float* __restrict__ T, const float* __restrict__ dinv,
                                 float* __restrict__ Agg, int n) {
  int i = blockIdx.x * blockDim.x + threadIdx.x;
  if (i < n) {
    float d = dinv[i >> 7];
    Agg[i] = T[i] * d * d;
  }
}

// ---------------- edge messages: Agg[dst] += T[src] * dinv[src]*dinv[dst] ----------------
__global__ void edge_msg_kernel(const int* __restrict__ src, const int* __restrict__ dst,
                                const float* __restrict__ dinv, const float* __restrict__ T,
                                float* __restrict__ Agg, int ne) {
  int e = blockIdx.x * (blockDim.x >> 5) + (threadIdx.x >> 5);
  int lane = threadIdx.x & 31;
  if (e >= ne) return;
  int s = src[e], d = dst[e];
  float norm = dinv[s] * dinv[d];
  float4 v = ((const float4*)(T + (size_t)s * GH))[lane];
  float* op = Agg + (size_t)d * GH + 4 * lane;
  atomicAdd(op + 0, v.x * norm);
  atomicAdd(op + 1, v.y * norm);
  atomicAdd(op + 2, v.z * norm);
  atomicAdd(op + 3, v.w * norm);
}

// ---------------- fused bias + eval-BN + ReLU -> bf16 activations ----------------
__global__ void bn_relu_kernel(const float* __restrict__ Agg, const float* __restrict__ b,
                               const float* __restrict__ g, const float* __restrict__ be,
                               const float* __restrict__ m, const float* __restrict__ v,
                               __bf16* __restrict__ Xb, int n) {
  int i = blockIdx.x * blockDim.x + threadIdx.x;
  if (i < n) {
    int f = i & (GH - 1);
    float h = Agg[i] + b[f];
    float y = (h - m[f]) * rsqrtf(v[f] + 1e-5f) * g[f] + be[f];
    Xb[i] = (__bf16)fmaxf(y, 0.0f);
  }
}

// ---------------- final: Out(N,10) = Xb(N,128) @ Wpo(packed) + bo ----------------
__global__ void gemm_out_wmma(const __bf16* __restrict__ Xb, const __bf16* __restrict__ Wpo,
                              const float* __restrict__ bo, float* __restrict__ Out) {
  const int wave = threadIdx.x >> 5;
  const int lane = threadIdx.x & 31;
  const int m    = lane & 15;
  const int half = lane >> 4;
  const int rt   = blockIdx.x * 8 + wave;
  if (rt >= GN / 16) return;           // wave-uniform exit keeps EXEC all-1s for WMMA

  const __bf16* xr0 = Xb + (size_t)(rt * 16 + m) * GH;
  const __bf16* wp0 = Wpo + (size_t)m * 32 + half * 16;
  v8f acc = {};
#pragma unroll
  for (int kt = 0; kt < 4; ++kt) {
    const __bf16* xr = xr0 + kt * 32 + half * 16;
    v8bf alo = *(const v8bf*)xr;
    v8bf ahi = *(const v8bf*)(xr + 8);
    v16bf a;
#pragma unroll
    for (int e = 0; e < 8; ++e) { a[e] = alo[e]; a[e + 8] = ahi[e]; }
    v16bf b = *(const v16bf*)(wp0 + (size_t)kt * 512);
    acc = __builtin_amdgcn_wmma_f32_16x16x32_bf16(false, a, false, b,
                                                  (short)0, acc, false, false);
  }
  if (m < GC) {
    float bias = bo[m];
#pragma unroll
    for (int g = 0; g < 8; ++g)
      Out[(size_t)(rt * 16 + half * 8 + g) * GC + m] = acc[g] + bias;
  }
}

extern "C" void kernel_launch(void* const* d_in, const int* in_sizes, int n_in,
                              void* d_out, int out_size, void* d_ws, size_t ws_size,
                              hipStream_t stream) {
  const float* x   = (const float*)d_in[0];
  const int*   ei  = (const int*)d_in[1];
  const int*   src = ei;
  const int*   dst = ei + GE;
  const float* w[4];  const float* bb[4]; const float* gg[4];
  const float* be[4]; const float* mm[4]; const float* vv[4];
  for (int l = 0; l < 4; ++l) {
    int base = 2 + 6 * l;
    w[l]  = (const float*)d_in[base + 0];
    bb[l] = (const float*)d_in[base + 1];
    gg[l] = (const float*)d_in[base + 2];
    be[l] = (const float*)d_in[base + 3];
    mm[l] = (const float*)d_in[base + 4];
    vv[l] = (const float*)d_in[base + 5];
  }
  const float* w_out = (const float*)d_in[26];
  const float* b_out = (const float*)d_in[27];
  float* out = (float*)d_out;

  // workspace: deg[N] f32 | dinv[N] f32 | T[N*128] f32 | Agg[N*128] f32 |
  //            Xb[N*128] bf16 | Wp[4*16384] bf16 | Wpo[2048] bf16
  char* p = (char*)d_ws;
  auto alloc = [&](size_t bytes) {
    char* r = p;
    p += (bytes + 255) & ~(size_t)255;
    return r;
  };
  float*  deg  = (float*)alloc((size_t)GN * 4);
  float*  dinv = (float*)alloc((size_t)GN * 4);
  float*  T    = (float*)alloc((size_t)GN * GH * 4);
  float*  Agg  = (float*)alloc((size_t)GN * GH * 4);
  __bf16* Xb   = (__bf16*)alloc((size_t)GN * GH * 2);
  __bf16* Wp   = (__bf16*)alloc((size_t)4 * GH * GH * 2);
  __bf16* Wpo  = (__bf16*)alloc((size_t)2048 * 2);

  const int NT = 256;
  const int nNF = GN * GH;

  // one-time packs / converts (re-run every launch: deterministic, graph-safe)
  for (int l = 0; l < 4; ++l)
    pack_w_kernel<<<(GH * GH) / NT, NT, 0, stream>>>(w[l], Wp + (size_t)l * GH * GH);
  pack_wo_kernel<<<2048 / NT, NT, 0, stream>>>(w_out, Wpo);
  convert_x_kernel<<<(nNF + NT - 1) / NT, NT, 0, stream>>>(x, Xb, nNF);

  // degrees -> dinv
  zero_f32<<<(GN + NT - 1) / NT, NT, 0, stream>>>(deg, GN);
  degree_kernel<<<(GE + NT - 1) / NT, NT, 0, stream>>>(dst, deg, GE);
  dinv_kernel<<<(GN + NT - 1) / NT, NT, 0, stream>>>(deg, dinv, GN);

  const int edge_blocks = (GE + (NT / 32) - 1) / (NT / 32);
  for (int l = 0; l < 4; ++l) {
    gemm_wmma<<<GN / 16, NT, 0, stream>>>(Xb, Wp + (size_t)l * GH * GH, T);
    self_loop_kernel<<<(nNF + NT - 1) / NT, NT, 0, stream>>>(T, dinv, Agg, nNF);
    edge_msg_kernel<<<edge_blocks, NT, 0, stream>>>(src, dst, dinv, T, Agg, GE);
    bn_relu_kernel<<<(nNF + NT - 1) / NT, NT, 0, stream>>>(Agg, bb[l], gg[l], be[l],
                                                           mm[l], vv[l], Xb, nNF);
  }
  gemm_out_wmma<<<(GN / 16 + 7) / 8, NT, 0, stream>>>(Xb, Wpo, b_out, out);
}
